// LightningIndexer_82686710383010
// MI455X (gfx1250) — matrix operand order repaired and checked by reading, not compiled, not run on hardware
//
#include <hip/hip_runtime.h>

typedef __attribute__((ext_vector_type(16))) __bf16 bf16x16;
typedef __attribute__((ext_vector_type(8)))  __bf16 bf16x8;
typedef __attribute__((ext_vector_type(8)))  float  f32x8;
typedef __attribute__((ext_vector_type(4)))  float  f32x4;

static constexpr int BN = 2;        // batch
static constexpr int T  = 4096;     // sequence
static constexpr int C  = 2048;     // n_embd
static constexpr int H  = 16;       // indexer heads
static constexpr int DH = 32;       // indexer dim (== WMMA bf16 K depth!)
static constexpr int M  = BN * T;   // 8192 flattened rows

// ---------------------------------------------------------------------------
// f32 -> bf16 conversion, 8 elements per thread (all sizes are multiples of 8).
// Inputs are read exactly once -> non-temporal loads (TH=NT) so the 67 MB x
// tensor does not displace anything useful in L2.
// ---------------------------------------------------------------------------
__global__ void cvt_f32_to_bf16(const float* __restrict__ in,
                                __bf16* __restrict__ out, int n) {
  int i = (blockIdx.x * blockDim.x + threadIdx.x) * 8;
  if (i + 8 <= n) {
    f32x4 a = __builtin_nontemporal_load((const f32x4*)(in + i));
    f32x4 b = __builtin_nontemporal_load((const f32x4*)(in + i + 4));
    bf16x8 o;
#pragma unroll
    for (int j = 0; j < 4; ++j) { o[j] = (__bf16)a[j]; o[j + 4] = (__bf16)b[j]; }
    *(bf16x8*)(out + i) = o;   // re-read by GEMMs -> regular temporal store
  }
}

// ---------------------------------------------------------------------------
// Fragment loaders per CDNA5 ISA 7.12.2 (wave32):
// A (16x32 bf16, MxK): lane l -> row = l&15; K-chunk base kb = (l>=16)*8;
//   elems 0..7  = K[kb .. kb+7], elems 8..15 = K[kb+16 .. kb+23]  (contiguous)
// B (32x16 bf16, KxN): lane l -> col n = l&15; kb = (l>=16)*8;
//   elems 0..7  = K[kb .. kb+7], elems 8..15 = K[kb+16 .. kb+23]  (same as A)
// ---------------------------------------------------------------------------
__device__ __forceinline__ bf16x16 load_a_frag(const __bf16* base, int ld, int lane) {
  const __bf16* p = base + (size_t)(lane & 15) * ld + ((lane >> 4) << 3);
  union { bf16x16 v; bf16x8 h[2]; } u;
  u.h[0] = *(const bf16x8*)(p);
  u.h[1] = *(const bf16x8*)(p + 16);
  return u.v;
}

__device__ __forceinline__ bf16x16 load_b_frag(const __bf16* base, int ld, int lane) {
  const __bf16* p = base + (size_t)(lane & 15) * ld + ((lane >> 4) << 3);
  union { bf16x16 v; bf16x8 h[2]; } u;
  u.h[0] = *(const bf16x8*)(p);
  u.h[1] = *(const bf16x8*)(p + 16);
  return u.v;
}

// ---------------------------------------------------------------------------
// GEMM: Out(M,N) bf16 = A(M,K=2048) bf16 * W(N,K=2048)^T  (one 16x16 tile/wave)
// Used for the Q projection (N=512) and the K projection (N=32).
// ---------------------------------------------------------------------------
__global__ void gemm_bf16(const __bf16* __restrict__ A, const __bf16* __restrict__ W,
                          __bf16* __restrict__ Out, int N, int NT) {
  int wave = (int)((blockIdx.x * blockDim.x + threadIdx.x) >> 5);
  int lane = threadIdx.x & 31;
  int mt = wave / NT, nt = wave % NT;
  const __bf16* Ab = A + (size_t)(mt * 16) * C;
  const __bf16* Wb = W + (size_t)(nt * 16) * C;
  f32x8 acc = {};
  for (int k0 = 0; k0 < C; k0 += 32) {
    bf16x16 a = load_a_frag(Ab + k0, C, lane);
    bf16x16 b = load_b_frag(Wb + k0, C, lane);
    acc = __builtin_amdgcn_wmma_f32_16x16x32_bf16(false, a, false, b,
                                                  (short)0, acc, false, false);
  }
  // C/D layout: VGPR r, lane l -> row = r + (l>=16 ? 8 : 0), col = l&15
  int roff = (lane >> 4) << 3;
  int col  = nt * 16 + (lane & 15);
#pragma unroll
  for (int r = 0; r < 8; ++r)
    Out[(size_t)(mt * 16 + roff + r) * N + col] = (__bf16)acc[r];
}

// ---------------------------------------------------------------------------
// w projection: Wf(B,H,T) f32 = x(M,2048) * Ww(16,2048)^T, stored transposed
// so the score kernel reads 8 contiguous per-row weights as one v8f.
// One wave per 16-row tile (512 waves).
// ---------------------------------------------------------------------------
__global__ void gemm_wproj(const __bf16* __restrict__ A, const __bf16* __restrict__ W,
                           float* __restrict__ Out) {
  int wave = (int)((blockIdx.x * blockDim.x + threadIdx.x) >> 5);
  int lane = threadIdx.x & 31;
  const __bf16* Ab = A + (size_t)(wave * 16) * C;
  f32x8 acc = {};
  for (int k0 = 0; k0 < C; k0 += 32) {
    bf16x16 a = load_a_frag(Ab + k0, C, lane);
    bf16x16 b = load_b_frag(W + k0, C, lane);
    acc = __builtin_amdgcn_wmma_f32_16x16x32_bf16(false, a, false, b,
                                                  (short)0, acc, false, false);
  }
  int roff = (lane >> 4) << 3;
  int h    = lane & 15;
  int row0 = wave * 16;           // tile never straddles a batch (4096 % 16 == 0)
  int b    = row0 / T;
#pragma unroll
  for (int r = 0; r < 8; ++r) {
    int t = row0 + roff + r - b * T;
    Out[(size_t)(b * H + h) * T + t] = acc[r];
  }
}

// ---------------------------------------------------------------------------
// Score kernel: importance[b,q,k] = sum_h w[b,q,h] * relu( q[b,q,h,:]·k[b,k,:] )
// Each wave: one 16(q) x 64(k) tile. 4 K-fragments loaded once and reused
// across 16 heads; one v_wmma_f32_16x16x32_bf16 per (head, key-tile).
// Output is 134 MB write-once -> non-temporal stores keep the hot Q/K/Wf
// read set (9 MB) resident in the 192 MB L2 for the whole pass.
// ---------------------------------------------------------------------------
__global__ void scores_kernel(const __bf16* __restrict__ Q, const __bf16* __restrict__ K,
                              const float* __restrict__ Wf, float* __restrict__ Out) {
  int wave = (int)((blockIdx.x * blockDim.x + threadIdx.x) >> 5);
  int lane = threadIdx.x & 31;
  int kg = wave & 63;            // 64-key group
  int qt = (wave >> 6) & 255;    // 16-query tile
  int b  = wave >> 14;           // batch
  int qrow0 = qt * 16;
  int key0  = kg * 64;
  int roff  = (lane >> 4) << 3;

  // B-operand fragments: K rows (key position = n), d = 8h..8h+7, 16+8h..16+8h+7 per lane
  bf16x16 bf[4];
#pragma unroll
  for (int kt = 0; kt < 4; ++kt) {
    const __bf16* p = K + (size_t)(b * T + key0 + kt * 16 + (lane & 15)) * DH
                        + ((lane >> 4) << 3);
    union { bf16x16 v; bf16x8 h[2]; } u;
    u.h[0] = *(const bf16x8*)(p);
    u.h[1] = *(const bf16x8*)(p + 16);
    bf[kt] = u.v;
  }

  f32x8 acc[4] = {};
  for (int h = 0; h < H; ++h) {
    // A-operand: Q[b, qrow, h, 0..31]; row stride between queries is H*DH
    const __bf16* pa = Q + ((size_t)(b * T + qrow0 + (lane & 15)) * H + h) * DH
                         + ((lane >> 4) << 3);
    union { bf16x16 v; bf16x8 c[2]; } au;
    au.c[0] = *(const bf16x8*)(pa);
    au.c[1] = *(const bf16x8*)(pa + 16);
    // head weights for this lane's 8 accumulator rows (contiguous in (B,H,T))
    f32x8 wv = *(const f32x8*)(Wf + (size_t)(b * H + h) * T + qrow0 + roff);
    f32x8 zero = {};
#pragma unroll
    for (int kt = 0; kt < 4; ++kt) {
      f32x8 c = __builtin_amdgcn_wmma_f32_16x16x32_bf16(false, au.v, false, bf[kt],
                                                        (short)0, zero, false, false);
#pragma unroll
      for (int r = 0; r < 8; ++r)
        acc[kt][r] += wv[r] * fmaxf(c[r], 0.0f);
    }
  }

  int col0 = key0 + (lane & 15);
#pragma unroll
  for (int kt = 0; kt < 4; ++kt)
#pragma unroll
    for (int r = 0; r < 8; ++r)
      __builtin_nontemporal_store(
          acc[kt][r],
          Out + (size_t)(b * T + qrow0 + roff + r) * T + col0 + kt * 16);
}

// ---------------------------------------------------------------------------
extern "C" void kernel_launch(void* const* d_in, const int* in_sizes, int n_in,
                              void* d_out, int out_size, void* d_ws, size_t ws_size,
                              hipStream_t stream) {
  const float* x  = (const float*)d_in[0];   // (2,4096,2048)
  const float* Wq = (const float*)d_in[1];   // (512,2048)
  const float* Wk = (const float*)d_in[2];   // (32,2048)
  const float* Ww = (const float*)d_in[3];   // (16,2048)
  float* out = (float*)d_out;                // (2,4096,4096)

  // workspace carve-up (all offsets 32B-aligned): ~45.3 MB total
  __bf16* xb  = (__bf16*)d_ws;                    // 8192*2048
  __bf16* Wqb = xb  + (size_t)M * C;              // 512*2048
  __bf16* Wkb = Wqb + (size_t)512 * C;            // 32*2048
  __bf16* Wwb = Wkb + (size_t)32 * C;             // 16*2048
  __bf16* Qb  = Wwb + (size_t)16 * C;             // 8192*512 bf16
  __bf16* Kb  = Qb  + (size_t)M * (H * DH);       // 8192*32  bf16
  float*  Wf  = (float*)(Kb + (size_t)M * DH);    // 2*16*4096 f32 (B,H,T)

  // 1) bf16 conversions
  cvt_f32_to_bf16<<<(M * C) / (256 * 8), 256, 0, stream>>>(x,  xb,  M * C);
  cvt_f32_to_bf16<<<(512 * C) / (256 * 8), 256, 0, stream>>>(Wq, Wqb, 512 * C);
  cvt_f32_to_bf16<<<(32 * C) / (256 * 8), 256, 0, stream>>>(Wk, Wkb, 32 * C);
  cvt_f32_to_bf16<<<(16 * C) / (256 * 8), 256, 0, stream>>>(Ww, Wwb, 16 * C);

  // 2) projections via WMMA (one 16x16 tile per wave, 8 waves/block)
  //    Q: 512 Mtiles * 32 Ntiles = 16384 waves -> 2048 blocks
  gemm_bf16<<<2048, 256, 0, stream>>>(xb, Wqb, Qb, H * DH, (H * DH) / 16);
  //    K: 512 * 2 = 1024 waves -> 128 blocks
  gemm_bf16<<<128, 256, 0, stream>>>(xb, Wkb, Kb, DH, DH / 16);
  //    w: 512 waves -> 64 blocks, transposed f32 output
  gemm_wproj<<<64, 256, 0, stream>>>(xb, Wwb, Wf);

  // 3) fused scores + relu + head-weighted reduction
  //    2 * 256 qtiles * 64 kgroups = 32768 waves -> 4096 blocks
  scores_kernel<<<4096, 256, 0, stream>>>(Qb, Kb, Wf, out);
}
